// LatentSpace_44976897524406
// MI455X (gfx1250) — matrix operand order repaired and checked by reading, not compiled, not run on hardware
//
#include <hip/hip_runtime.h>
#include <math.h>

typedef __attribute__((ext_vector_type(16))) _Float16 v16h;
typedef __attribute__((ext_vector_type(8)))  float    v8f;

#define NCLUS   16
#define NCAT    136         // 16*17/2 upper-tri pairs (incl. diagonal)
#define DIM     64
#define TILE_M  16
#define WAVES_PER_BLOCK 8
#define SAMPLES_PER_BLOCK (TILE_M * WAVES_PER_BLOCK)   // 128

__global__ __launch_bounds__(256)
void latent_logpz_kernel(const float* __restrict__ z,   // [S, 64] row-major
                         const float* __restrict__ pi,  // [136]
                         const float* __restrict__ mu,  // [64, 16] row-major
                         float* __restrict__ blockPartial)
{
    // ---------------- shared staging ----------------
    __shared__ float sMnorm[NCLUS];
    __shared__ int   sA[NCAT], sB[NCAT];
    // per-category record (32B): [invsig, recip, amB, mnB, logclip, sqrtc, logpi, ab-pack]
    __shared__ __align__(16) float sParams[NCAT][8];
    __shared__ __align__(32) _Float16 sBpack[2 * 32 * 16]; // [chunk][lane][16 halves]
    __shared__ float sW[WAVES_PER_BLOCK][TILE_M][NCLUS + 1]; // +1 pad vs bank conflicts
    __shared__ float sZsqp[WAVES_PER_BLOCK][32];
    __shared__ float sWaveSum[WAVES_PER_BLOCK];

    const int t = threadIdx.x;
    const float log2pi = 1.8378770664093453f;

    // ---- phase 0: cluster norms, pair indices, B-matrix f16 pack ----
    if (t < NCLUS) {
        float acc = 0.0f;
        #pragma unroll 8
        for (int d = 0; d < DIM; ++d) {
            float v = mu[d * NCLUS + t];
            acc += v * v;
        }
        sMnorm[t] = acc;
    }
    if (t < NCAT) {
        // c-th upper-tri pair in row-major order: (0,0)..(0,15),(1,1),...
        int rem = t, a = 0;
        while (rem >= (NCLUS - a)) { rem -= (NCLUS - a); ++a; }
        sA[t] = a;
        sB[t] = a + rem;
    }
    {
        // sBpack: B operand for v_wmma_f32_16x16x32_f16 (32x16 f16), contiguous k order matching A:
        // VGPR i, lane<16: K={2i,2i+1}, col=lane; lane>=16: K={16+2i,16+2i+1}, col=lane-16
        int lane = (t >> 3) & 31;
        int i    = t & 7;
        int col  = lane & 15;
        int koff = (lane >> 4) ? 16 : 0;
        #pragma unroll
        for (int q = 0; q < 2; ++q) {
            int K = q * 32 + koff + 2 * i;
            sBpack[(q * 32 + lane) * 16 + 2 * i]     = (_Float16)mu[K * NCLUS + col];
            sBpack[(q * 32 + lane) * 16 + 2 * i + 1] = (_Float16)mu[(K + 1) * NCLUS + col];
        }
    }
    __syncthreads();

    // ---- phase 1a: per-category constants (threads 0..135) ----
    if (t < NCAT) {
        int a = sA[t], b = sB[t];
        float cross = 0.0f;
        #pragma unroll 8
        for (int d = 0; d < DIM; ++d)
            cross += mu[d * NCLUS + a] * mu[d * NCLUS + b];
        float invsig = (a == b) ? 0.0f : (sMnorm[a] + sMnorm[b] - 2.0f * cross);
        float clip_is = fminf(fmaxf(invsig, 1e-12f), 1e30f);

        // log softmax(pi) with clip — redundant per-thread scan of 136 (tiny)
        float pmax = -1e30f;
        for (int c = 0; c < NCAT; ++c) pmax = fmaxf(pmax, pi[c]);
        float psum = 0.0f;
        for (int c = 0; c < NCAT; ++c) psum += __expf(pi[c] - pmax);
        float p = __expf(pi[t] - pmax) / psum;
        p = fminf(fmaxf(p, 1e-16f), 1.0f);

        sParams[t][0] = invsig;
        sParams[t][1] = (invsig == 0.0f) ? 0.0f : 1.0f / invsig;
        sParams[t][2] = sMnorm[b] - cross;           // dot(alpha, mu_B)
        sParams[t][3] = sMnorm[b];
        sParams[t][4] = __logf(clip_is);
        sParams[t][5] = sqrtf(clip_is);
        sParams[t][6] = __logf(p);
        ((int*)sParams[t])[7] = a | (b << 8);        // packed indices
    }

    // ---- phase 1b: WMMA  W = z_tile @ mu  (f16 in, f32 acc) ----
    const int wave = t >> 5;
    const int lane = t & 31;
    const int col  = lane & 15;       // row within tile for A; N col for B/D
    const int half = lane >> 4;
    const int kb   = half ? 16 : 0;

    const int tile = blockIdx.x * WAVES_PER_BLOCK + wave;
    const float* zrow = z + (size_t)(tile * TILE_M + col) * DIM;

    v16h aChunk[2];
    float ss = 0.0f;
    #pragma unroll
    for (int q = 0; q < 2; ++q) {
        #pragma unroll
        for (int i = 0; i < 8; ++i) {
            int K = q * 32 + kb + 2 * i;
            float x0 = zrow[K];
            float x1 = zrow[K + 1];
            ss += x0 * x0 + x1 * x1;
            aChunk[q][2 * i]     = (_Float16)x0;
            aChunk[q][2 * i + 1] = (_Float16)x1;
        }
    }
    sZsqp[wave][lane] = ss;

    v16h b0 = *(const v16h*)&sBpack[(0 * 32 + lane) * 16];
    v16h b1 = *(const v16h*)&sBpack[(1 * 32 + lane) * 16];

    v8f cacc = {};
    cacc = __builtin_amdgcn_wmma_f32_16x16x32_f16(false, aChunk[0], false, b0,
                                                  (short)0, cacc, false, false);
    cacc = __builtin_amdgcn_wmma_f32_16x16x32_f16(false, aChunk[1], false, b1,
                                                  (short)0, cacc, false, false);

    // D layout: lane<16 -> (M=r, N=lane); lane>=16 -> (M=r+8, N=lane-16)
    #pragma unroll
    for (int r = 0; r < 8; ++r)
        sW[wave][r + 8 * half][col] = cacc[r];

    __syncthreads();

    // ---- phase 2: per-sample category sweep + online logsumexp ----
    const int s = col;                                // my sample within the tile
    const float* sWrow = &sW[wave][s][0];             // dynamic LDS reads, no reg-index
    const float zsq = sZsqp[wave][s] + sZsqp[wave][s + 16];
    const float baseC = -0.5f * (float)DIM * log2pi;

    float m = -INFINITY, S = 0.0f;
    const int c0 = half * (NCAT / 2);                 // 68 cats per lane-half
    for (int i = 0; i < NCAT / 2; ++i) {
        int c = c0 + i;
        const float4* pp = (const float4*)&sParams[c][0];
        float4 p0 = pp[0];                            // invsig, recip, amB, mnB
        float4 p1 = pp[1];                            // logclip, sqrtc, logpi, ab
        int ab = __float_as_int(p1.w);
        float wa = sWrow[ab & 0xff];
        float wb = sWrow[(ab >> 8) & 0xff];

        float beta_sq = zsq - 2.0f * wb + p0.w;
        float dab = (wb - wa) - p0.z;                 // dot(alpha, beta)
        float nu  = -dab * p0.y;
        float tq  = fmaf(nu * nu, p0.x, -beta_sq);
        float log_eta0 = 0.5f * (log2pi - p1.x + tq);
        float cd  = normcdff((1.0f - nu) * p1.y) - normcdff(-nu * p1.y);
        cd = fminf(fmaxf(cd, 1e-16f), 1e30f);
        float gen = log_eta0 + __logf(cd);

        float val = (p0.x == 0.0f) ? (-0.5f * beta_sq) : gen;   // diag select
        val += baseC + p1.z;

        // branchless online logsumexp
        float nm = fmaxf(m, val);
        S = S * __expf(m - nm) + __expf(val - nm);
        m = nm;
    }

    // combine the two lane-halves of each sample
    float mo = __shfl_xor(m, 16, 32);
    float So = __shfl_xor(S, 16, 32);
    float M  = fmaxf(m, mo);
    float lse = M + __logf(S * __expf(m - M) + So * __expf(mo - M));

    // deterministic wave reduction over the 16 samples (zero out mirror half)
    float v = (half == 0) ? lse : 0.0f;
    #pragma unroll
    for (int mask = 1; mask < 32; mask <<= 1)
        v += __shfl_xor(v, mask, 32);

    if (lane == 0) sWaveSum[wave] = v;
    __syncthreads();
    if (t == 0) {
        float acc = 0.0f;
        #pragma unroll
        for (int wv = 0; wv < WAVES_PER_BLOCK; ++wv) acc += sWaveSum[wv];
        blockPartial[blockIdx.x] = acc;
    }
}

__global__ void latent_finalize_kernel(const float* __restrict__ blockPartial,
                                       float* __restrict__ out,
                                       int nblocks, float invN)
{
    if (threadIdx.x == 0) {
        float s = 0.0f;
        for (int i = 0; i < nblocks; ++i) s += blockPartial[i]; // fixed order: deterministic
        out[0] = s * invN;
    }
}

extern "C" void kernel_launch(void* const* d_in, const int* in_sizes, int n_in,
                              void* d_out, int out_size, void* d_ws, size_t ws_size,
                              hipStream_t stream) {
    const float* z  = (const float*)d_in[0];   // [512*16, 64]
    const float* pi = (const float*)d_in[1];   // [136]
    const float* mu = (const float*)d_in[2];   // [64, 16]
    float* out = (float*)d_out;
    float* partial = (float*)d_ws;             // one float per block

    int nSamples = in_sizes[0] / DIM;                       // 8192
    int nBlocks  = nSamples / SAMPLES_PER_BLOCK;            // 64

    latent_logpz_kernel<<<nBlocks, 256, 0, stream>>>(z, pi, mu, partial);
    latent_finalize_kernel<<<1, 32, 0, stream>>>(partial, out, nBlocks,
                                                 1.0f / (float)nSamples);
}
